// Attention_35862976921921
// MI455X (gfx1250) — matrix operand order repaired and checked by reading, not compiled, not run on hardware
//
#include <hip/hip_runtime.h>

#define B_  2
#define S_  2048
#define D_  4096
#define H_  32
#define HD_ 128

typedef __bf16 bf16;
typedef __attribute__((ext_vector_type(16))) __bf16 v16bf;
typedef __attribute__((ext_vector_type(8)))  __bf16 v8bf;
typedef __attribute__((ext_vector_type(8)))  float  v8f;

#if defined(__has_builtin)
#if __has_builtin(__builtin_amdgcn_global_load_async_to_lds_b128) && \
    __has_builtin(__builtin_amdgcn_s_wait_asynccnt)
#define HAVE_ASYNC_LDS 1
#endif
#endif

#ifdef HAVE_ASYNC_LDS
// Builtin operates on int4 pointers: (global src, LDS dst, imm offset, imm cpol)
typedef int v4i_ __attribute__((vector_size(16)));
typedef __attribute__((address_space(1))) v4i_ gv4i;
typedef __attribute__((address_space(3))) v4i_ sv4i;

// 16B/lane global -> LDS, no VGPR data path; tracked by ASYNCcnt.
static __device__ __forceinline__ void async_cp16(bf16* lds, const bf16* g) {
  __builtin_amdgcn_global_load_async_to_lds_b128((gv4i*)g, (sv4i*)lds, 0, 0);
}
static __device__ __forceinline__ void wait_async0() {
  __builtin_amdgcn_s_wait_asynccnt(0);
}
#endif

static __device__ __forceinline__ v16bf frag2(const bf16* p0, const bf16* p1) {
  v8bf lo = *(const v8bf*)p0;
  v8bf hi = *(const v8bf*)p1;
  return __builtin_shufflevector(lo, hi, 0,1,2,3,4,5,6,7,8,9,10,11,12,13,14,15);
}

static __device__ __forceinline__ v8f wmma_bf16(v16bf a, v16bf b, v8f c) {
  // (neg_a, A, neg_b, B, c_mod, C, reuse_a, reuse_b)
  return __builtin_amdgcn_wmma_f32_16x16x32_bf16(false, a, false, b, (short)0, c, false, false);
}

static __device__ __forceinline__ v8f vzero8() {
  v8f z = {0.f,0.f,0.f,0.f,0.f,0.f,0.f,0.f};
  return z;
}

// ---------------------------------------------------------------- cvt
__global__ void cvt_f32_to_bf16(const float* __restrict__ s, bf16* __restrict__ d, int n) {
  int i = blockIdx.x * blockDim.x + threadIdx.x;
  int st = gridDim.x * blockDim.x;
  for (; i < n; i += st) d[i] = (bf16)s[i];
}

// ---------------------------------------------------------------- GEMM
// Y[m][n] = sum_k A[m][k] * W[n][k]   (M = N = K = 4096)
// MODE 0: RoPE -> Qbuf [b][h][s][hd]
// MODE 1: RoPE -> Kbuf [b][h][s][hd]
// MODE 2:       -> Vt  [b][h][hd][s]
// MODE 3: f32   -> out [m][n]
#define LDS_STRIDE 40
template<int MODE>
__global__ __launch_bounds__(256) void gemm_wmma(
    const bf16* __restrict__ A, const bf16* __restrict__ W,
    bf16* __restrict__ outb, float* __restrict__ outf,
    const float* __restrict__ cosT, const float* __restrict__ sinT)
{
  const int tid  = threadIdx.x;
  const int lane = tid & 31;
  const int w    = tid >> 5;     // 0..7
  const int wm   = w >> 1;       // 0..3  (32-row strip)
  const int wn   = w & 1;        // 0..1  (64-col strip)
  const int ln   = lane & 15;
  const int hi   = lane >> 4;    // 0/1
  const int m0   = blockIdx.y * 128;
  const int n0   = blockIdx.x * 128;

  const int ldr = tid >> 1;          // staging row 0..127
  const int ldc = (tid & 1) * 16;    // staging col 0 / 16

  v8f acc[2][4];
#pragma unroll
  for (int i = 0; i < 2; i++)
#pragma unroll
    for (int j = 0; j < 4; j++) acc[i][j] = vzero8();

  const int akb = hi * 8;    // A frag: lanes16-31 start at K=8
  const int bkb = hi * 8;    // B frag: lanes16-31 start at K=8

#ifdef HAVE_ASYNC_LDS
  // -------- async double-buffered pipeline: one barrier per K-step --------
  __shared__ bf16 lA[2][128 * LDS_STRIDE];
  __shared__ bf16 lB[2][128 * LDS_STRIDE];

  const bf16* ga = A + (size_t)(m0 + ldr) * 4096 + ldc;
  const bf16* gb = W + (size_t)(n0 + ldr) * 4096 + ldc;
  const int soff = ldr * LDS_STRIDE + ldc;

  async_cp16(&lA[0][soff], ga);      async_cp16(&lA[0][soff + 8], ga + 8);
  async_cp16(&lB[0][soff], gb);      async_cp16(&lB[0][soff + 8], gb + 8);

  for (int k0 = 0; k0 < 4096; k0 += 32) {
    const int cur = (k0 >> 5) & 1;
    wait_async0();         // my async copies into buf[cur] are in LDS
    __syncthreads();       // everyone's copies visible; prev readers done

    if (k0 + 32 < 4096) {  // kick off next tile into the other buffer
      const bf16* pa = ga + k0 + 32;
      const bf16* pb = gb + k0 + 32;
      async_cp16(&lA[cur ^ 1][soff], pa);  async_cp16(&lA[cur ^ 1][soff + 8], pa + 8);
      async_cp16(&lB[cur ^ 1][soff], pb);  async_cp16(&lB[cur ^ 1][soff + 8], pb + 8);
    }

    v16bf af[2], bfr[4];
#pragma unroll
    for (int mi = 0; mi < 2; mi++) {
      const bf16* p = &lA[cur][(wm * 32 + mi * 16 + ln) * LDS_STRIDE + akb];
      af[mi] = frag2(p, p + 16);
    }
#pragma unroll
    for (int ni = 0; ni < 4; ni++) {
      const bf16* p = &lB[cur][(wn * 64 + ni * 16 + ln) * LDS_STRIDE + bkb];
      bfr[ni] = frag2(p, p + 16);
    }
#pragma unroll
    for (int mi = 0; mi < 2; mi++)
#pragma unroll
      for (int ni = 0; ni < 4; ni++)
        acc[mi][ni] = wmma_bf16(af[mi], bfr[ni], acc[mi][ni]);
  }
#else
  // -------- fallback: register-prefetch pipeline (round-1 path) --------
  __shared__ bf16 lA[128 * LDS_STRIDE];
  __shared__ bf16 lB[128 * LDS_STRIDE];

  v8bf ra0, ra1, rb0, rb1;
  {
    const bf16* pa = A + (size_t)(m0 + ldr) * 4096 + ldc;
    const bf16* pb = W + (size_t)(n0 + ldr) * 4096 + ldc;
    ra0 = *(const v8bf*)pa;  ra1 = *(const v8bf*)(pa + 8);
    rb0 = *(const v8bf*)pb;  rb1 = *(const v8bf*)(pb + 8);
  }
  for (int k0 = 0; k0 < 4096; k0 += 32) {
    __syncthreads();
    *(v8bf*)(lA + ldr * LDS_STRIDE + ldc)     = ra0;
    *(v8bf*)(lA + ldr * LDS_STRIDE + ldc + 8) = ra1;
    *(v8bf*)(lB + ldr * LDS_STRIDE + ldc)     = rb0;
    *(v8bf*)(lB + ldr * LDS_STRIDE + ldc + 8) = rb1;
    __syncthreads();
    if (k0 + 32 < 4096) {
      const bf16* pa = A + (size_t)(m0 + ldr) * 4096 + k0 + 32 + ldc;
      const bf16* pb = W + (size_t)(n0 + ldr) * 4096 + k0 + 32 + ldc;
      ra0 = *(const v8bf*)pa;  ra1 = *(const v8bf*)(pa + 8);
      rb0 = *(const v8bf*)pb;  rb1 = *(const v8bf*)(pb + 8);
    }
    v16bf af[2], bfr[4];
#pragma unroll
    for (int mi = 0; mi < 2; mi++) {
      const bf16* p = lA + (wm * 32 + mi * 16 + ln) * LDS_STRIDE + akb;
      af[mi] = frag2(p, p + 16);
    }
#pragma unroll
    for (int ni = 0; ni < 4; ni++) {
      const bf16* p = lB + (wn * 64 + ni * 16 + ln) * LDS_STRIDE + bkb;
      bfr[ni] = frag2(p, p + 16);
    }
#pragma unroll
    for (int mi = 0; mi < 2; mi++)
#pragma unroll
      for (int ni = 0; ni < 4; ni++)
        acc[mi][ni] = wmma_bf16(af[mi], bfr[ni], acc[mi][ni]);
  }
#endif

  // ---- epilogue ----
#pragma unroll
  for (int mi = 0; mi < 2; mi++)
#pragma unroll
    for (int ni = 0; ni < 4; ni++)
#pragma unroll
      for (int r = 0; r < 8; r++) {
        const int m = m0 + wm * 32 + mi * 16 + hi * 8 + r;
        const int n = n0 + wn * 64 + ni * 16 + ln;
        float v = acc[mi][ni][r];
        if (MODE == 3) {
          outf[(size_t)m * 4096 + n] = v;
        } else if (MODE == 2) {
          const int b = m >> 11, s = m & (S_ - 1);
          const int h = n >> 7,  hd = n & (HD_ - 1);
          outb[(((size_t)b * H_ + h) * HD_ + hd) * S_ + s] = (bf16)v;
        } else {  // MODE 0/1: RoPE.  Adjacent columns live in adjacent lanes.
          const float vp = __shfl_xor(v, 1, 32);
          const int s  = m & (S_ - 1);
          const int hd = n & (HD_ - 1);
          const int fi = hd >> 1;
          const float c  = cosT[s * (HD_ / 2) + fi];
          const float sn = sinT[s * (HD_ / 2) + fi];
          const float o = ((hd & 1) == 0) ? (v * c - vp * sn) : (vp * sn + v * c);
          const int b = m >> 11, h = n >> 7;
          outb[(((size_t)b * H_ + h) * S_ + s) * HD_ + hd] = (bf16)o;
        }
      }
}

// ---------------------------------------------------------------- attention
// grid: (S/128, H, B), 256 threads = 8 waves; wave w owns q-rows [w*16, w*16+16)
#define AT_STRIDE 136
__global__ __launch_bounds__(256) void attn_wmma(
    const bf16* __restrict__ Q, const bf16* __restrict__ K,
    const bf16* __restrict__ Vt, bf16* __restrict__ O)
{
  __shared__ bf16 lK[128 * AT_STRIDE];   // [key][d]
  __shared__ bf16 lV[128 * AT_STRIDE];   // [d][key]   (V transposed)
  __shared__ bf16 lP[128 * AT_STRIDE];   // [qrow][key]

  const int tid  = threadIdx.x;
  const int lane = tid & 31;
  const int w    = tid >> 5;
  const int ln   = lane & 15;
  const int hi   = lane >> 4;
  const int qb   = blockIdx.x;
  const int h    = blockIdx.y;
  const int b    = blockIdx.z;
  const int q0   = qb * 128;
  const size_t head = ((size_t)b * H_ + h) * (size_t)S_ * HD_;

  // Q fragments straight from global (A layout = two 16B halves per lane)
  v16bf qf[4];
  {
    const bf16* qp = Q + head + (size_t)(q0 + w * 16 + ln) * HD_;
#pragma unroll
    for (int ks = 0; ks < 4; ks++) {
      const bf16* p = qp + ks * 32 + hi * 8;
      qf[ks] = frag2(p, p + 16);
    }
  }

  v8f  o[8];
  float mrow[8], lrow[8];
#pragma unroll
  for (int nt = 0; nt < 8; nt++) o[nt] = vzero8();
#pragma unroll
  for (int r = 0; r < 8; r++) { mrow[r] = -3.0e38f; lrow[r] = 0.f; }

  const int ldr = tid >> 1;          // 0..127
  const int ldc = (tid & 1) * 64;    // 0 / 64
  const float sc = 0.08838834764831845f;   // 1/sqrt(128)

  for (int kb = 0; kb <= qb; kb++) {
    { // stage K tile [128 keys x 128 d] and Vt tile [128 d x 128 keys]
      const bf16* kg = K  + head + (size_t)(kb * 128 + ldr) * HD_ + ldc;
      const bf16* vg = Vt + head + (size_t)ldr * S_ + kb * 128 + ldc;
      bf16* ks_ = lK + ldr * AT_STRIDE + ldc;
      bf16* vs_ = lV + ldr * AT_STRIDE + ldc;
#ifdef HAVE_ASYNC_LDS
#pragma unroll
      for (int j = 0; j < 8; j++) {
        async_cp16(ks_ + j * 8, kg + j * 8);
        async_cp16(vs_ + j * 8, vg + j * 8);
      }
      wait_async0();
#else
#pragma unroll
      for (int j = 0; j < 8; j++) {
        *(v8bf*)(ks_ + j * 8) = *(const v8bf*)(kg + j * 8);
        *(v8bf*)(vs_ + j * 8) = *(const v8bf*)(vg + j * 8);
      }
#endif
      if (kb < qb) {  // warm WGP$/L2 for the next tiles while we compute
        __builtin_prefetch(kg + (size_t)128 * HD_, 0, 3);
        __builtin_prefetch(vg + 128, 0, 3);
      }
    }
    __syncthreads();

    // ---- S = Q K^T ----
    v8f st[8];
#pragma unroll
    for (int nt = 0; nt < 8; nt++) st[nt] = vzero8();
#pragma unroll
    for (int ks = 0; ks < 4; ks++) {
      const int kof = ks * 32 + hi * 8;
#pragma unroll
      for (int nt = 0; nt < 8; nt++) {
        const bf16* p = lK + (nt * 16 + ln) * AT_STRIDE + kof;
        st[nt] = wmma_bf16(qf[ks], frag2(p, p + 16), st[nt]);
      }
    }

    // ---- scale + causal mask + row max ----
    float cm[8];
#pragma unroll
    for (int r = 0; r < 8; r++) cm[r] = -3.0e38f;
#pragma unroll
    for (int nt = 0; nt < 8; nt++)
#pragma unroll
      for (int r = 0; r < 8; r++) {
        float sv = st[nt][r] * sc;
        if (kb == qb) {
          const int kg = kb * 128 + nt * 16 + ln;
          const int qg = q0 + w * 16 + hi * 8 + r;
          if (kg > qg) sv = -1.0e30f;
        }
        st[nt][r] = sv;
        cm[r] = fmaxf(cm[r], sv);
      }
#pragma unroll
    for (int r = 0; r < 8; r++) {   // rows live inside a 16-lane half
      cm[r] = fmaxf(cm[r], __shfl_xor(cm[r], 1, 32));
      cm[r] = fmaxf(cm[r], __shfl_xor(cm[r], 2, 32));
      cm[r] = fmaxf(cm[r], __shfl_xor(cm[r], 4, 32));
      cm[r] = fmaxf(cm[r], __shfl_xor(cm[r], 8, 32));
    }
    float alpha[8];
#pragma unroll
    for (int r = 0; r < 8; r++) {
      const float mn = fmaxf(mrow[r], cm[r]);
      alpha[r] = __expf(mrow[r] - mn);
      mrow[r] = mn;
    }

    // ---- P = exp(S - m), to LDS (bf16), row sums ----
    float rs[8];
#pragma unroll
    for (int r = 0; r < 8; r++) rs[r] = 0.f;
#pragma unroll
    for (int nt = 0; nt < 8; nt++)
#pragma unroll
      for (int r = 0; r < 8; r++) {
        const float p = __expf(st[nt][r] - mrow[r]);
        rs[r] += p;
        lP[(w * 16 + hi * 8 + r) * AT_STRIDE + nt * 16 + ln] = (bf16)p;
      }
#pragma unroll
    for (int r = 0; r < 8; r++) {
      rs[r] += __shfl_xor(rs[r], 1, 32);
      rs[r] += __shfl_xor(rs[r], 2, 32);
      rs[r] += __shfl_xor(rs[r], 4, 32);
      rs[r] += __shfl_xor(rs[r], 8, 32);
      lrow[r] = lrow[r] * alpha[r] + rs[r];
    }
#pragma unroll
    for (int nt = 0; nt < 8; nt++)
#pragma unroll
      for (int r = 0; r < 8; r++) o[nt][r] = o[nt][r] * alpha[r];

    // ---- O += P V  (A frags from own wave's lP rows; B frags from lV) ----
#pragma unroll
    for (int ks = 0; ks < 4; ks++) {
      const bf16* ap = lP + (w * 16 + ln) * AT_STRIDE + ks * 32 + hi * 8;
      const v16bf afr = frag2(ap, ap + 16);
#pragma unroll
      for (int nt = 0; nt < 8; nt++) {
        const bf16* bp = lV + (nt * 16 + ln) * AT_STRIDE + ks * 32 + hi * 8;
        o[nt] = wmma_bf16(afr, frag2(bp, bp + 16), o[nt]);
      }
    }
    __syncthreads();   // protect lK/lV before next tile load
  }

  // ---- normalize and write O in [b][s][h*HD+d] layout for the out-proj GEMM
#pragma unroll
  for (int nt = 0; nt < 8; nt++)
#pragma unroll
    for (int r = 0; r < 8; r++) {
      const int row = q0 + w * 16 + hi * 8 + r;
      const int col = h * HD_ + nt * 16 + ln;
      O[((size_t)b * S_ + row) * D_ + col] = (bf16)(o[nt][r] * (1.0f / lrow[r]));
    }
}

// ---------------------------------------------------------------- launch
extern "C" void kernel_launch(void* const* d_in, const int* in_sizes, int n_in,
                              void* d_out, int out_size, void* d_ws, size_t ws_size,
                              hipStream_t stream) {
  const float* x  = (const float*)d_in[0];
  const float* fc = (const float*)d_in[1];   // (S, HD/2)
  const float* fs = (const float*)d_in[2];
  // d_in[3] = mask (causal, reproduced analytically)
  const float* wq = (const float*)d_in[4];
  const float* wk = (const float*)d_in[5];
  const float* wv = (const float*)d_in[6];
  const float* wo = (const float*)d_in[7];
  // d_in[8], d_in[9] = zero caches, d_in[10] = start_pos (=0): unused
  float* out = (float*)d_out;

  const size_t NE = (size_t)4096 * 4096;   // == B*S*D == D*D
  bf16* ws  = (bf16*)d_ws;
  bf16* Xb  = ws;
  bf16* Wqb = ws + 1 * NE;
  bf16* Wkb = ws + 2 * NE;
  bf16* Wvb = ws + 3 * NE;
  bf16* Wob = ws + 4 * NE;
  bf16* Qb  = ws + 5 * NE;
  bf16* Kb  = ws + 6 * NE;
  bf16* Vtb = ws + 7 * NE;
  bf16* Ob  = ws + 8 * NE;

  cvt_f32_to_bf16<<<2048, 256, 0, stream>>>(x,  Xb,  (int)NE);
  cvt_f32_to_bf16<<<2048, 256, 0, stream>>>(wq, Wqb, (int)NE);
  cvt_f32_to_bf16<<<2048, 256, 0, stream>>>(wk, Wkb, (int)NE);
  cvt_f32_to_bf16<<<2048, 256, 0, stream>>>(wv, Wvb, (int)NE);
  cvt_f32_to_bf16<<<2048, 256, 0, stream>>>(wo, Wob, (int)NE);

  dim3 g(32, 32), blk(256);
  gemm_wmma<0><<<g, blk, 0, stream>>>(Xb, Wqb, Qb,  nullptr, fc, fs);
  gemm_wmma<1><<<g, blk, 0, stream>>>(Xb, Wkb, Kb,  nullptr, fc, fs);
  gemm_wmma<2><<<g, blk, 0, stream>>>(Xb, Wvb, Vtb, nullptr, nullptr, nullptr);

  attn_wmma<<<dim3(S_ / 128, H_, B_), 256, 0, stream>>>(Qb, Kb, Vtb, Ob);

  gemm_wmma<3><<<g, blk, 0, stream>>>(Ob, Wob, nullptr, out, nullptr, nullptr);
}
